// TreeLSTMModel_63239098466675
// MI455X (gfx1250) — compile-verified, matrix operand order checked by reading, not run on hardware
//
#include <hip/hip_runtime.h>
#include <hip/hip_bf16.h>
#include <math.h>

typedef __attribute__((ext_vector_type(16))) _Float16 v16h;
typedef __attribute__((ext_vector_type(8)))  _Float16 v8h;
typedef __attribute__((ext_vector_type(8)))  float    v8f;
typedef __attribute__((ext_vector_type(4)))  float    v4f;

#define TREES   256
#define S_NODES 255
#define NTOT    (TREES * S_NODES)   // 65280
#define HDIM    128
#define FIN     64
#define KCAT    192                 // [features(64) ; h(128)]

__device__ __forceinline__ float sigm(float x){ return 1.0f/(1.0f+expf(-x)); }

// ---------------------------------------------------------------- prep ------
__global__ void cvt_feat_kernel(const float* __restrict__ f, _Float16* __restrict__ o, int n){
  int i = blockIdx.x*blockDim.x + threadIdx.x;
  if (i < n) o[i] = (_Float16)f[i];
}

// Pack BT[j][k] = (k<64 ? W[k][j] : U[k-64][j]) as f16, row j is K-contiguous.
__global__ void pack_wu_kernel(const float* __restrict__ W, const float* __restrict__ U,
                               _Float16* __restrict__ BT, int ncol){
  int j = blockIdx.x;          // output column 0..ncol-1
  int k = threadIdx.x;         // 0..191
  float v = (k < FIN) ? W[k*ncol + j] : U[(k-FIN)*ncol + j];
  BT[j*KCAT + k] = (_Float16)v;
}

// ---------------------------------------------------------------- GEMM ------
// mode 0: leaves iou : A = featH[g]                (K=64)  -> iou[r*384+col]
// mode 1: inner  iou : A = [featH[g]; h[cl]+h[cr]] (K=192) -> iou[r*384+col]
// mode 2: forget f   : A = [featH[parent]; h[child]] (K=192)
//                      -> fc[r*128+col] = sigmoid(acc) * c[child][col]
// Block = 4 waves; each wave computes a 16x64 output slab (4 N-tiles) so one
// A fragment feeds 4 independent v_wmma_f32_16x16x32_f16 accumulators.
__global__ __launch_bounds__(128) void tree_gemm_kernel(
    int mode, int L,
    const _Float16* __restrict__ featH,
    const float*    __restrict__ hbuf,
    const float*    __restrict__ cbuf,
    const _Float16* __restrict__ BT,
    const float*    __restrict__ bias,
    float*          __restrict__ outbuf,
    int nChunks, int outStride)
{
  const int wave = threadIdx.x >> 5;
  const int lane = threadIdx.x & 31;
  const int rr   = lane & 15;
  const int half = lane >> 4;

  // A-row mapping (tree index math; adjacency is a perfect binary tree)
  const int mtile = blockIdx.x*4 + wave;
  const int r     = mtile*16 + rr;
  const int m     = (mode == 2) ? (r >> 1) : r;
  const int lmask = (1 << L) - 1;
  const int t     = m >> L;
  const int local = lmask + (m & lmask);
  const int g     = t*S_NODES + local;           // node (mode 2: parent)
  const int cl    = t*S_NODES + 2*local + 1;     // left child
  const int gch   = cl + (r & 1);                // mode-2 child row
  const _Float16* frow = featH + (size_t)g*FIN;
  const float*    hl   = hbuf + (size_t)cl*HDIM;       // mode 1 left child
  const float*    hr   = hbuf + (size_t)(cl+1)*HDIM;   // mode 1 right child
  const float*    hc   = hbuf + (size_t)gch*HDIM;      // mode 2 child

  // B columns: 4 N-tiles per wave
  const int ncolBase = blockIdx.y*64;
  const _Float16* brow = BT + (size_t)(ncolBase + rr)*KCAT;   // tile 0; +16 rows per tile

  v8f acc[4];
  #pragma unroll
  for (int nt = 0; nt < 4; ++nt) {
    float bv = bias[ncolBase + nt*16 + rr];
    #pragma unroll
    for (int i = 0; i < 8; ++i) acc[nt][i] = bv;
  }

  for (int kc = 0; kc < nChunks; ++kc) {
    const int kbase = kc*32;
    v16h afrag;

    // ---- A fragment: elems 0..7 <- K kbase+half*8, elems 8..15 <- +16 ----
    if (kbase < FIN) {                 // feature part of K (already f16)
      const _Float16* p  = frow + kbase + half*8;
      v8h a0 = *(const v8h*)p;
      v8h a1 = *(const v8h*)(p + 16);
      #pragma unroll
      for (int i = 0; i < 8; ++i){ afrag[i] = a0[i]; afrag[8+i] = a1[i]; }
    } else {                           // recurrent part of K (f32 h -> f16)
      const int ko = kbase - FIN + half*8;
      if (mode == 1) {                 // sum of the two children, fused
        const float* pl = hl + ko;
        const float* pr = hr + ko;
        v4f l0=*(const v4f*)pl, l1=*(const v4f*)(pl+4), l2=*(const v4f*)(pl+16), l3=*(const v4f*)(pl+20);
        v4f r0=*(const v4f*)pr, r1=*(const v4f*)(pr+4), r2=*(const v4f*)(pr+16), r3=*(const v4f*)(pr+20);
        #pragma unroll
        for (int i = 0; i < 4; ++i){
          afrag[i]    = (_Float16)(l0[i]+r0[i]);
          afrag[4+i]  = (_Float16)(l1[i]+r1[i]);
          afrag[8+i]  = (_Float16)(l2[i]+r2[i]);
          afrag[12+i] = (_Float16)(l3[i]+r3[i]);
        }
        if (kc + 1 < nChunks) {        // gfx1250 global_prefetch_b8 path
          __builtin_prefetch(pl + 32, 0, 1);
          __builtin_prefetch(pr + 32, 0, 1);
        }
      } else {                         // mode 2: single child row
        const float* p = hc + ko;
        v4f a0=*(const v4f*)p, a1=*(const v4f*)(p+4), a2=*(const v4f*)(p+16), a3=*(const v4f*)(p+20);
        #pragma unroll
        for (int i = 0; i < 4; ++i){
          afrag[i]    = (_Float16)a0[i];
          afrag[4+i]  = (_Float16)a1[i];
          afrag[8+i]  = (_Float16)a2[i];
          afrag[12+i] = (_Float16)a3[i];
        }
        if (kc + 1 < nChunks) __builtin_prefetch(p + 32, 0, 1);
      }
    }

    // ---- 4 B fragments, 4 independent accumulators ----
    #pragma unroll
    for (int nt = 0; nt < 4; ++nt) {
      const _Float16* q  = brow + (size_t)(nt*16)*KCAT + kbase + half*8;
      v8h b0 = *(const v8h*)q;
      v8h b1 = *(const v8h*)(q + 16);
      v16h bfrag;
      #pragma unroll
      for (int i = 0; i < 8; ++i){ bfrag[i] = b0[i]; bfrag[8+i] = b1[i]; }
      acc[nt] = __builtin_amdgcn_wmma_f32_16x16x32_f16(false, afrag, false, bfrag,
                                                       (short)0, acc[nt], false, false);
    }
  }

  // Epilogue. C/D layout: lane -> col (within tile), VGPR v -> row v + 8*half.
  #pragma unroll
  for (int nt = 0; nt < 4; ++nt) {
    const int col = ncolBase + nt*16 + rr;
    #pragma unroll
    for (int v = 0; v < 8; ++v) {
      const int orow = mtile*16 + v + 8*half;
      float val = acc[nt][v];
      if (mode <= 1) {
        outbuf[(size_t)orow*outStride + col] = val;
      } else {
        const int mm  = orow >> 1;
        const int tt  = mm >> L;
        const int loc = lmask + (mm & lmask);
        const int gc  = tt*S_NODES + 2*loc + 1 + (orow & 1);
        outbuf[(size_t)orow*outStride + col] = sigm(val) * cbuf[(size_t)gc*HDIM + col];
      }
    }
  }
}

// ---------------------------------------------------------------- cell ------
__global__ void lstm_cell_kernel(int L, int leaves,
                                 const float* __restrict__ iou,
                                 const float* __restrict__ fc,
                                 float* __restrict__ h, float* __restrict__ c)
{
  const int m = blockIdx.x;
  const int j = threadIdx.x;                  // 0..127
  const int lmask = (1 << L) - 1;
  const int t = m >> L;
  const int local = lmask + (m & lmask);
  const int g = t*S_NODES + local;

  float iv = iou[(size_t)m*384 + j];
  float ov = iou[(size_t)m*384 + 128 + j];
  float uv = iou[(size_t)m*384 + 256 + j];
  float csum = 0.0f;
  if (!leaves)
    csum = fc[(size_t)(2*m)*HDIM + j] + fc[(size_t)(2*m + 1)*HDIM + j];
  float cn = sigm(iv)*tanhf(uv) + csum;
  float hn = sigm(ov)*tanhf(cn);
  h[(size_t)g*HDIM + j] = hn;
  c[(size_t)g*HDIM + j] = cn;
}

// ---------------------------------------------------------------- head ------
__global__ void tree_mean_kernel(const float* __restrict__ h, float* __restrict__ x){
  const int t = blockIdx.x, j = threadIdx.x;
  const float* base = h + (size_t)t*S_NODES*HDIM + j;
  float s = 0.0f;
  for (int n = 0; n < S_NODES; ++n) s += base[(size_t)n*HDIM];
  float v = s * (1.0f/255.0f);
  x[t*HDIM + j] = v > 0.0f ? v : 0.0f;
}

__global__ void mlp_kernel(const float* __restrict__ x,
                           const float* __restrict__ w1, const float* __restrict__ b1,
                           const float* __restrict__ w2, const float* __restrict__ b2,
                           const float* __restrict__ w3, const float* __restrict__ b3,
                           float* __restrict__ out)
{
  __shared__ float s0[HDIM];
  __shared__ float s1[HDIM];
  const int t = blockIdx.x, j = threadIdx.x;
  s0[j] = x[t*HDIM + j];
  __syncthreads();
  float a = b1[j];
  for (int k = 0; k < HDIM; ++k) a += s0[k] * w1[k*HDIM + j];
  a = a > 0.0f ? a : 0.0f;
  s1[j] = a;
  __syncthreads();
  a = b2[j];
  for (int k = 0; k < HDIM; ++k) a += s1[k] * w2[k*HDIM + j];
  a = a > 0.0f ? a : 0.0f;
  s0[j] = a * w3[j];
  __syncthreads();
  if (j == 0) {
    float s = b3[0];
    for (int k = 0; k < HDIM; ++k) s += s0[k];
    out[t] = s;
  }
}

// ---------------------------------------------------------------- launch ----
extern "C" void kernel_launch(void* const* d_in, const int* in_sizes, int n_in,
                              void* d_out, int out_size, void* d_ws, size_t ws_size,
                              hipStream_t stream)
{
  (void)in_sizes; (void)n_in; (void)out_size; (void)ws_size;
  const float* features = (const float*)d_in[0];
  const float* W_iou    = (const float*)d_in[1];
  const float* b_iou    = (const float*)d_in[2];
  const float* U_iou    = (const float*)d_in[3];
  const float* W_f      = (const float*)d_in[4];
  const float* b_f      = (const float*)d_in[5];
  const float* U_f      = (const float*)d_in[6];
  const float* lin1_w   = (const float*)d_in[7];
  const float* lin1_b   = (const float*)d_in[8];
  const float* lin2_w   = (const float*)d_in[9];
  const float* lin2_b   = (const float*)d_in[10];
  const float* lin_w    = (const float*)d_in[11];
  const float* lin_b    = (const float*)d_in[12];
  float* out = (float*)d_out;

  char* ws = (char*)d_ws;
  size_t off = 0;
  auto take = [&](size_t bytes)->char*{
    char* p = ws + off;
    off = (off + bytes + 255) & ~(size_t)255;
    return p;
  };
  _Float16* featH = (_Float16*)take((size_t)NTOT*FIN*2);       //  8.4 MB
  _Float16* WUiou = (_Float16*)take((size_t)384*KCAT*2);       //  0.15 MB
  _Float16* WUf   = (_Float16*)take((size_t)128*KCAT*2);       //  0.05 MB
  float* hbuf = (float*)take((size_t)NTOT*HDIM*4);             // 33.4 MB
  float* cbuf = (float*)take((size_t)NTOT*HDIM*4);             // 33.4 MB
  float* ioub = (float*)take((size_t)32768*384*4);             // 50.3 MB (max level M)
  float* fcb  = (float*)take((size_t)32768*HDIM*4);            // 16.8 MB
  float* xb   = (float*)take((size_t)TREES*HDIM*4);            //  0.13 MB

  // one-time conversions / weight packing
  cvt_feat_kernel<<<(NTOT*FIN + 255)/256, 256, 0, stream>>>(features, featH, NTOT*FIN);
  pack_wu_kernel<<<384, KCAT, 0, stream>>>(W_iou, U_iou, WUiou, 384);
  pack_wu_kernel<<<128, KCAT, 0, stream>>>(W_f,   U_f,   WUf,   128);

  // leaves (tree level L=7): iou = features @ W_iou + b_iou, c_sum = 0
  {
    const int M = TREES << 7;   // 32768
    tree_gemm_kernel<<<dim3(M/64, 384/64), 128, 0, stream>>>(
        0, 7, featH, hbuf, cbuf, WUiou, b_iou, ioub, /*K=64*/2, 384);
    lstm_cell_kernel<<<M, HDIM, 0, stream>>>(7, 1, ioub, fcb, hbuf, cbuf);
  }

  // internal levels, root last
  for (int L = 6; L >= 0; --L) {
    const int M = TREES << L;
    tree_gemm_kernel<<<dim3(M/64, 384/64), 128, 0, stream>>>(
        1, L, featH, hbuf, cbuf, WUiou, b_iou, ioub, /*K=192*/6, 384);
    tree_gemm_kernel<<<dim3((2*M)/64, 128/64), 128, 0, stream>>>(
        2, L, featH, hbuf, cbuf, WUf, b_f, fcb, /*K=192*/6, 128);
    lstm_cell_kernel<<<M, HDIM, 0, stream>>>(L, 0, ioub, fcb, hbuf, cbuf);
  }

  // readout
  tree_mean_kernel<<<TREES, HDIM, 0, stream>>>(hbuf, xb);
  mlp_kernel<<<TREES, HDIM, 0, stream>>>(xb, lin1_w, lin1_b, lin2_w, lin2_b,
                                         lin_w, lin_b, out);
}
